// MultiScaleDeformableAlingment_37572373905603
// MI455X (gfx1250) — compile-verified, matrix operand order checked by reading, not run on hardware
//
#include <hip/hip_runtime.h>

typedef __attribute__((ext_vector_type(16))) _Float16 v16h;
typedef __attribute__((ext_vector_type(8)))  float    v8f;

#define TPB 256
#define TILE_PIX 64

// ---------------- weight packing (f32 -> f16, zero-pad M to mult of 16) ----------------
__global__ void pack_w_kernel(const float* __restrict__ w, _Float16* __restrict__ wtr,
                              int Cout, int Mpad, int K) {
  int i = blockIdx.x * blockDim.x + threadIdx.x;
  if (i >= Mpad * K) return;
  int co = i / K, r = i % K;
  wtr[i] = (co < Cout) ? (_Float16)w[(size_t)co * K + r] : (_Float16)0.f;
}

// deform weights: wtr[o][(g*9+k)*16 + c] = w[o][(g*16+c)][k]   (w: 64x64x3x3)
__global__ void pack_w_deform_kernel(const float* __restrict__ w, _Float16* __restrict__ wtr) {
  int i = blockIdx.x * blockDim.x + threadIdx.x;
  if (i >= 64 * 576) return;
  int o = i / 576, r = i % 576;
  int gk = r >> 4, c = r & 15;
  int g = gk / 9, k = gk % 9;
  wtr[i] = (_Float16)w[(size_t)o * 576 + (g * 16 + c) * 9 + k];
}

// ---------------- fragment load: 16 halfs = two 16B chunks per lane ----------------
__device__ __forceinline__ v16h ldfrag(const _Float16* row, int hi) {
  v16h f;
#pragma unroll
  for (int j = 0; j < 8; ++j) {
    f[j]     = row[hi * 8 + j];
    f[j + 8] = row[16 + hi * 8 + j];
  }
  return f;
}

__device__ __forceinline__ float sigmoid_fast(float v) {
  return __builtin_amdgcn_rcpf(1.f + __expf(-v));   // v_exp_f32 + v_rcp_f32
}

// ---------------- generic 3x3 conv as implicit GEMM (templated: K compile-time) ----------
// PAIR=1: image n -> (b=n/4, g=n%4); ci<16 from sou group, ci>=16 from ref group.
// EPI=1: offset map  rng*2*sigmoid(v)-rng -> out0[(b*72 + g*18 + co)*HW + p]
// EPI=2: modulator   2*sigmoid(v)         -> out0[(n*36 + co)*HW + p]
template <int CIN, int PAIR, int EPI>
__global__ __launch_bounds__(TPB) void conv3x3_wmma(
    const float* __restrict__ srcA, const float* __restrict__ srcB,
    int H, int W,
    const _Float16* __restrict__ wtr, const float* __restrict__ bias,
    int Cout, int Mtiles, float* __restrict__ out0, float rng) {
  extern __shared__ _Float16 patch[];
  constexpr int K = CIN * 9;
  constexpr int KS = K + 8;          // halfs; rows stay 16B-aligned
  constexpr int KSTEPS = K / 32;
  const int HW = H * W;
  const int n     = blockIdx.y;
  const int ptile = blockIdx.x * TILE_PIX;
  const int tid   = threadIdx.x;

  // ---- im2col into LDS (f16), reduce index r = ci*9 + k; exact multiple of TPB ----
  for (int t = tid; t < TILE_PIX * CIN; t += TPB) {
    int pl = t % TILE_PIX;
    int ci = t / TILE_PIX;
    int p = ptile + pl;
    int y = p / W, x = p % W;
    const float* src;
    if (PAIR) {
      int b = n >> 2, g = n & 3;
      int cc = ci & 15;
      const float* base = (ci < 16) ? srcA : srcB;
      src = base + ((size_t)b * 64 + g * 16 + cc) * HW;
    } else {
      src = srcA + ((size_t)n * CIN + ci) * HW;
    }
    _Float16* dst = patch + pl * KS + ci * 9;
#pragma unroll
    for (int kk = 0; kk < 9; ++kk) {
      int yy = y - 1 + kk / 3, xx = x - 1 + kk % 3;
      float v = (yy >= 0 && yy < H && xx >= 0 && xx < W) ? src[yy * W + xx] : 0.f;
      dst[kk] = (_Float16)v;
    }
  }
  __syncthreads();

  // ---- WMMA: each wave-task = one M tile x two adjacent N tiles (A reused) ----
  const int wave = tid >> 5;
  const int lane = tid & 31;
  const int l = lane & 15, hi = lane >> 4;
  for (int task = wave; task < Mtiles * 2; task += 8) {
    int mtile = task >> 1, np = task & 1;
    const _Float16* arow  = wtr + (size_t)(mtile * 16 + l) * K;
    const _Float16* brow0 = patch + (np * 32 + l) * KS;
    const _Float16* brow1 = brow0 + 16 * KS;
    v8f c0 = {0.f, 0.f, 0.f, 0.f, 0.f, 0.f, 0.f, 0.f};
    v8f c1 = {0.f, 0.f, 0.f, 0.f, 0.f, 0.f, 0.f, 0.f};
#pragma unroll
    for (int s = 0; s < KSTEPS; ++s) {
      v16h a  = ldfrag(arow + s * 32, hi);
      v16h b0 = ldfrag(brow0 + s * 32, hi);
      v16h b1 = ldfrag(brow1 + s * 32, hi);
      c0 = __builtin_amdgcn_wmma_f32_16x16x32_f16(false, a, false, b0, (short)0, c0, false, false);
      c1 = __builtin_amdgcn_wmma_f32_16x16x32_f16(false, a, false, b1, (short)0, c1, false, false);
    }
    // epilogue for both N tiles
#pragma unroll
    for (int nt = 0; nt < 2; ++nt) {
      const v8f& c = nt ? c1 : c0;
      int p = ptile + np * 32 + nt * 16 + l;
#pragma unroll
      for (int r = 0; r < 8; ++r) {
        int co = mtile * 16 + r + hi * 8;
        if (co < Cout) {
          float v = c[r] + bias[co];
          if (EPI == 1) {
            int b = n >> 2, g = n & 3;
            out0[((size_t)b * 72 + g * 18 + co) * HW + p] =
                rng * 2.f * sigmoid_fast(v) - rng;
          } else {
            out0[((size_t)n * 36 + co) * HW + p] = 2.f * sigmoid_fast(v);
          }
        }
      }
    }
  }
}

// ---------------- fused deformable sampling + WMMA GEMM ----------------
__global__ __launch_bounds__(TPB) void deform_wmma(
    const float* __restrict__ sou, const float* __restrict__ offmap,
    const float* __restrict__ modmap, const _Float16* __restrict__ wtr,
    float* __restrict__ out, int H, int W) {
  extern __shared__ _Float16 patch[];
  constexpr int K = 576, KS = K + 8, KSTEPS = K / 32;
  const int HW = H * W;
  const int b     = blockIdx.y;
  const int ptile = blockIdx.x * TILE_PIX;
  const int tid   = threadIdx.x;

  // ---- sample stage: 64 pixels x (g,k)=36 tasks x 16 channels; 2304 = 9*256 tasks ----
  for (int t = tid; t < TILE_PIX * 36; t += TPB) {
    int pl = t % TILE_PIX;
    int gk = t / TILE_PIX;
    int g = gk / 9, k = gk % 9;
    int p = ptile + pl;
    int y = p / W, x = p % W;
    float dy = offmap[((size_t)b * 72 + g * 18 + k * 2) * HW + p];
    float dx = offmap[((size_t)b * 72 + g * 18 + k * 2 + 1) * HW + p];
    float m  = modmap[((size_t)b * 36 + g * 9 + k) * HW + p];
    float py = (float)(y - 1 + k / 3) + dy;
    float px = (float)(x - 1 + k % 3) + dx;
    float y0f = floorf(py), x0f = floorf(px);
    float wy = py - y0f, wx = px - x0f;
    int y0 = (int)y0f, x0 = (int)x0f;
    int y1 = y0 + 1, x1 = x0 + 1;
    float v00 = ((y0 >= 0 && y0 < H && x0 >= 0 && x0 < W) ? 1.f : 0.f) * (1.f - wy) * (1.f - wx);
    float v01 = ((y0 >= 0 && y0 < H && x1 >= 0 && x1 < W) ? 1.f : 0.f) * (1.f - wy) * wx;
    float v10 = ((y1 >= 0 && y1 < H && x0 >= 0 && x0 < W) ? 1.f : 0.f) * wy * (1.f - wx);
    float v11 = ((y1 >= 0 && y1 < H && x1 >= 0 && x1 < W) ? 1.f : 0.f) * wy * wx;
    int cy0 = y0 < 0 ? 0 : (y0 > H - 1 ? H - 1 : y0);
    int cy1 = y1 < 0 ? 0 : (y1 > H - 1 ? H - 1 : y1);
    int cx0 = x0 < 0 ? 0 : (x0 > W - 1 ? W - 1 : x0);
    int cx1 = x1 < 0 ? 0 : (x1 > W - 1 ? W - 1 : x1);
    const float* base = sou + ((size_t)b * 64 + g * 16) * HW;
    _Float16* dst = patch + pl * KS + gk * 16;
#pragma unroll 4
    for (int c = 0; c < 16; ++c) {
      const float* ch = base + (size_t)c * HW;
      float v = v00 * ch[cy0 * W + cx0] + v01 * ch[cy0 * W + cx1] +
                v10 * ch[cy1 * W + cx0] + v11 * ch[cy1 * W + cx1];
      dst[c] = (_Float16)(v * m);
    }
  }
  __syncthreads();

  // ---- GEMM: M=64 (4 tiles) x N=64 (4 tiles); wave = (mtile, N-tile pair) ----
  const int wave = tid >> 5;
  const int lane = tid & 31;
  const int l = lane & 15, hi = lane >> 4;
  const int mtile = wave >> 1, np = wave & 1;
  const _Float16* arow  = wtr + (size_t)(mtile * 16 + l) * K;
  const _Float16* brow0 = patch + (np * 32 + l) * KS;
  const _Float16* brow1 = brow0 + 16 * KS;
  v8f c0 = {0.f, 0.f, 0.f, 0.f, 0.f, 0.f, 0.f, 0.f};
  v8f c1 = {0.f, 0.f, 0.f, 0.f, 0.f, 0.f, 0.f, 0.f};
#pragma unroll
  for (int s = 0; s < KSTEPS; ++s) {
    v16h a  = ldfrag(arow + s * 32, hi);
    v16h b0 = ldfrag(brow0 + s * 32, hi);
    v16h b1 = ldfrag(brow1 + s * 32, hi);
    c0 = __builtin_amdgcn_wmma_f32_16x16x32_f16(false, a, false, b0, (short)0, c0, false, false);
    c1 = __builtin_amdgcn_wmma_f32_16x16x32_f16(false, a, false, b1, (short)0, c1, false, false);
  }
#pragma unroll
  for (int nt = 0; nt < 2; ++nt) {
    const v8f& c = nt ? c1 : c0;
    int p = ptile + np * 32 + nt * 16 + l;
#pragma unroll
    for (int r = 0; r < 8; ++r) {
      int co = mtile * 16 + r + hi * 8;
      out[((size_t)b * 64 + co) * HW + p] = c[r];
    }
  }
}

// ---------------- offset channel means (dy over 0::2, dx over 1::2) ----------------
__global__ void mean_off_kernel(const float* __restrict__ offmap, float* __restrict__ oyox, int HW) {
  int i = blockIdx.x * blockDim.x + threadIdx.x;
  if (i >= 2 * HW) return;
  int b = i / HW, p = i % HW;
  float sdy = 0.f, sdx = 0.f;
  for (int j = 0; j < 36; ++j) {
    sdy += offmap[((size_t)b * 72 + 2 * j) * HW + p];
    sdx += offmap[((size_t)b * 72 + 2 * j + 1) * HW + p];
  }
  oyox[((size_t)b * 2 + 0) * HW + p] = sdy * (1.f / 36.f);
  oyox[((size_t)b * 2 + 1) * HW + p] = sdx * (1.f / 36.f);
}

// ---------------- align_corners bilinear upsample to 512x512, scaled by fs ----------------
__global__ void upsample_ov_kernel(const float* __restrict__ oyox, float* __restrict__ out,
                                   int H, int W, float fs) {
  int i = blockIdx.x * blockDim.x + threadIdx.x;  // total = 2*2*512*512 = 2^20
  int xo = i & 511;
  int yo = (i >> 9) & 511;
  int ch = (i >> 18) & 1;
  int b  = i >> 19;
  float sy = (float)yo * (float)(H - 1) * (1.f / 511.f);
  float sx = (float)xo * (float)(W - 1) * (1.f / 511.f);
  int y0 = (int)floorf(sy); y0 = y0 < 0 ? 0 : (y0 > H - 2 ? H - 2 : y0);
  int x0 = (int)floorf(sx); x0 = x0 < 0 ? 0 : (x0 > W - 2 ? W - 2 : x0);
  float wy = sy - (float)y0, wx = sx - (float)x0;
  const float* pl = oyox + ((size_t)b * 2 + ch) * H * W;
  float v = pl[y0 * W + x0] * (1.f - wy) * (1.f - wx) +
            pl[y0 * W + x0 + 1] * (1.f - wy) * wx +
            pl[(y0 + 1) * W + x0] * wy * (1.f - wx) +
            pl[(y0 + 1) * W + x0 + 1] * wy * wx;
  out[i] = v * fs;
}

// ---------------- host driver ----------------
extern "C" void kernel_launch(void* const* d_in, const int* in_sizes, int n_in,
                              void* d_out, int out_size, void* d_ws, size_t ws_size,
                              hipStream_t stream) {
  (void)in_sizes; (void)n_in; (void)out_size; (void)ws_size;
  const float* sou[3]   = {(const float*)d_in[0],  (const float*)d_in[2],  (const float*)d_in[4]};
  const float* ref[3]   = {(const float*)d_in[1],  (const float*)d_in[3],  (const float*)d_in[5]};
  const float* off_w[3] = {(const float*)d_in[6],  (const float*)d_in[11], (const float*)d_in[16]};
  const float* off_b[3] = {(const float*)d_in[7],  (const float*)d_in[12], (const float*)d_in[17]};
  const float* mod_w[3] = {(const float*)d_in[8],  (const float*)d_in[13], (const float*)d_in[18]};
  const float* mod_b[3] = {(const float*)d_in[9],  (const float*)d_in[14], (const float*)d_in[19]};
  const float* reg_w[3] = {(const float*)d_in[10], (const float*)d_in[15], (const float*)d_in[20]};

  float* out = (float*)d_out;
  // output order: feat3, feat2, feat1, ov3, ov2, ov1
  float* featp[3] = {out + 655360, out + 131072, out + 0};   // scale idx 0->i=1,1->i=2,2->i=3
  float* ovp[3]   = {out + 4849664, out + 3801088, out + 2752512};
  const int   Hs[3]   = {128, 64, 32};
  const float rngs[3] = {32.f, 16.f, 8.f};   // min(H,W)*0.25
  const float fss[3]  = {4.f, 8.f, 16.f};

  size_t woff = 0;
  auto alloc = [&](size_t bytes) -> void* {
    void* p = (void*)((char*)d_ws + woff);
    woff += (bytes + 255) & ~(size_t)255;
    return p;
  };

  const size_t shm_off = (size_t)TILE_PIX * (32 * 9 + 8) * sizeof(_Float16);  // 37,888 B
  const size_t shm_big = (size_t)TILE_PIX * (64 * 9 + 8) * sizeof(_Float16);  // 74,752 B

  for (int s = 0; s < 3; ++s) {
    int H = Hs[s], W = Hs[s], HW = H * W;
    float*    offmap = (float*)alloc((size_t)2 * 72 * HW * sizeof(float));
    float*    modmap = (float*)alloc((size_t)2 * 36 * HW * sizeof(float));
    float*    oyox   = (float*)alloc((size_t)2 * 2 * HW * sizeof(float));
    _Float16* wtrO   = (_Float16*)alloc((size_t)32 * 288 * sizeof(_Float16));
    _Float16* wtrM   = (_Float16*)alloc((size_t)48 * 576 * sizeof(_Float16));
    _Float16* wtrR   = (_Float16*)alloc((size_t)64 * 576 * sizeof(_Float16));

    pack_w_kernel<<<(32 * 288 + TPB - 1) / TPB, TPB, 0, stream>>>(off_w[s], wtrO, 18, 32, 288);
    pack_w_kernel<<<(48 * 576 + TPB - 1) / TPB, TPB, 0, stream>>>(mod_w[s], wtrM, 36, 48, 576);
    pack_w_deform_kernel<<<(64 * 576 + TPB - 1) / TPB, TPB, 0, stream>>>(reg_w[s], wtrR);

    // offset conv: 8 images (b*G), Cin=32, Cout=18, Mtiles=2 -> sigmoid/range epilogue
    conv3x3_wmma<32, 1, 1><<<dim3(HW / TILE_PIX, 8), TPB, shm_off, stream>>>(
        sou[s], ref[s], H, W, wtrO, off_b[s], 18, 2, offmap, rngs[s]);
    // modulator conv: 2 images, Cin=64, Cout=36, Mtiles=3 -> 2*sigmoid epilogue
    conv3x3_wmma<64, 0, 2><<<dim3(HW / TILE_PIX, 2), TPB, shm_big, stream>>>(
        sou[s], nullptr, H, W, wtrM, mod_b[s], 36, 3, modmap, 0.f);
    // deformable conv: fused sample + GEMM -> feat output
    deform_wmma<<<dim3(HW / TILE_PIX, 2), TPB, shm_big, stream>>>(
        sou[s], offmap, modmap, wtrR, featp[s], H, W);
    // offset-vector outputs
    mean_off_kernel<<<(2 * HW + TPB - 1) / TPB, TPB, 0, stream>>>(offmap, oyox, HW);
    upsample_ov_kernel<<<(2 * 2 * 512 * 512) / TPB, TPB, 0, stream>>>(oyox, ovp[s], H, W, fss[s]);
  }
}
